// Model_1434519076876
// MI455X (gfx1250) — matrix operand order repaired and checked by reading, not compiled, not run on hardware
//
#include <hip/hip_runtime.h>
#include <hip/hip_bf16.h>

#define NNODES 101
#define NEGV -1e10f

typedef __attribute__((ext_vector_type(16))) __bf16 v16bf;
typedef __attribute__((ext_vector_type(8)))  __bf16 v8bf;
typedef __attribute__((ext_vector_type(8)))  float  v8f;

union ABPack { v16bf v; v8bf h[2]; };

__device__ __forceinline__ __bf16 to_bf16(float f) {
    union { float f; unsigned u; } in; in.f = f;
    unsigned u = in.u;
    u += 0x7fffu + ((u >> 16) & 1u);          // round-to-nearest-even
    union { unsigned short s; __bf16 b; } out;
    out.s = (unsigned short)(u >> 16);
    return out.b;
}

// ---------------------------------------------------------------------------
// WMMA GEMM: Out[M,N] = op(A)[M,K] @ Wt^T + bias, Wt stored [N][K] bf16.
// op(A): optional per-K-channel BN affine then ReLU, fp32 -> bf16 conversion.
// Block: 256 threads = 8 waves; block tile 128(M) x 128(N); K step 32.
// Double-buffered LDS tiles (1 barrier per k-step, global loads overlap WMMA).
// All 8 B fragments are loaded from LDS up front so the 8 WMMAs issue
// back-to-back behind a single s_wait_dscnt.
// ---------------------------------------------------------------------------
template<bool A_BF16, bool RELU_IN, bool HAS_AFFINE, bool OUT_RELU, bool OUT_BF16>
__global__ __launch_bounds__(256)
void gemm_wmma_bf16(const void* __restrict__ Aptr,
                    const __bf16* __restrict__ Wt,     // [N][K]
                    const float* __restrict__ bias,    // [N] or nullptr
                    const float* __restrict__ bn_g, const float* __restrict__ bn_b,
                    const float* __restrict__ bn_m, const float* __restrict__ bn_v,
                    void* __restrict__ Outptr, int M, int K, int N)
{
    __shared__ __bf16 lds_a[2][128][40];   // +8 pad keeps 16B alignment of chunks
    __shared__ __bf16 lds_b[2][128][40];

    const int tid  = threadIdx.x;
    const int wv   = tid >> 5;
    const int lane = tid & 31;
    const int r    = lane & 15;
    const int hi   = lane >> 4;

    const int rowBase = blockIdx.x * 128;
    const int colBase = blockIdx.y * 128;

    v8f acc[8] = {};

    const int mLoad = tid >> 1;         // 0..127
    const int kHalf = (tid & 1) * 16;   // 0 or 16

    uint4 aq0, aq1, bq0, bq1;           // staged tile fragments (registers)

    auto load_regs = [&](int ks) {
        const int row = rowBase + mLoad;
        const int kg0 = ks * 32 + kHalf;
        if (row < M) {
            if (A_BF16) {
                const uint4* src = (const uint4*)((const __bf16*)Aptr + (size_t)row * K + kg0);
                aq0 = src[0]; aq1 = src[1];
            } else {
                const float4* src = (const float4*)((const float*)Aptr + (size_t)row * K + kg0);
                union { __bf16 b[16]; uint4 q[2]; } av;
                #pragma unroll
                for (int j4 = 0; j4 < 4; ++j4) {
                    float4 f = src[j4];
                    float vs[4] = { f.x, f.y, f.z, f.w };
                    #pragma unroll
                    for (int u = 0; u < 4; ++u) {
                        float xv = vs[u];
                        if (HAS_AFFINE) {
                            const int kg = kg0 + j4 * 4 + u;
                            const float sc = bn_g[kg] * rsqrtf(bn_v[kg] + 1e-5f);
                            xv = xv * sc + (bn_b[kg] - bn_m[kg] * sc);
                        }
                        if (RELU_IN) xv = xv > 0.f ? xv : 0.f;
                        av.b[j4 * 4 + u] = to_bf16(xv);
                    }
                }
                aq0 = av.q[0]; aq1 = av.q[1];
            }
        } else {
            aq0 = make_uint4(0, 0, 0, 0);
            aq1 = make_uint4(0, 0, 0, 0);
        }
        const uint4* srcb = (const uint4*)(Wt + (size_t)(colBase + mLoad) * K + ks * 32 + kHalf);
        bq0 = srcb[0]; bq1 = srcb[1];
    };
    auto store_lds = [&](int buf) {
        *(uint4*)&lds_a[buf][mLoad][kHalf]     = aq0;
        *(uint4*)&lds_a[buf][mLoad][kHalf + 8] = aq1;
        *(uint4*)&lds_b[buf][mLoad][kHalf]     = bq0;
        *(uint4*)&lds_b[buf][mLoad][kHalf + 8] = bq1;
    };

    const int nSteps = K >> 5;
    load_regs(0);
    store_lds(0);
    __syncthreads();

    for (int ks = 0; ks < nSteps; ++ks) {
        const int cur  = ks & 1;
        const bool more = (ks + 1) < nSteps;
        if (more) load_regs(ks + 1);     // global loads overlap this tile's WMMAs

        // A layout (16-bit 16x32): lane=row r, elems 0..7 -> K=8*hi+j, 8..15 -> K=16+8*hi+j
        ABPack a;
        a.h[0] = *(const v8bf*)&lds_a[cur][16 * wv + r][8 * hi];
        a.h[1] = *(const v8bf*)&lds_a[cur][16 * wv + r][16 + 8 * hi];

        // B layout (32x16): lane=col r, elems 0..7 -> K=8*hi+j, 8..15 -> K=16+8*hi+j.
        // Load all 8 fragments first -> one dscnt wait, 8 back-to-back WMMAs.
        ABPack bfr[8];
        #pragma unroll
        for (int nt = 0; nt < 8; ++nt) {
            bfr[nt].h[0] = *(const v8bf*)&lds_b[cur][16 * nt + r][8 * hi];
            bfr[nt].h[1] = *(const v8bf*)&lds_b[cur][16 * nt + r][16 + 8 * hi];
        }
        #pragma unroll
        for (int nt = 0; nt < 8; ++nt) {
            acc[nt] = __builtin_amdgcn_wmma_f32_16x16x32_bf16(
                false, a.v, false, bfr[nt].v, (short)0, acc[nt], false, false);
        }

        if (more) store_lds(cur ^ 1);    // write the other buffer, then one barrier
        __syncthreads();
    }

    // ---- epilogue: D VGPR v -> row = v + 8*hi, col = r
    #pragma unroll
    for (int nt = 0; nt < 8; ++nt) {
        const int col = colBase + 16 * nt + r;
        const float bv = bias ? bias[col] : 0.f;
        #pragma unroll
        for (int v = 0; v < 8; ++v) {
            const int row = rowBase + 16 * wv + v + 8 * hi;
            if (row < M) {
                float o = acc[nt][v] + bv;
                if (OUT_RELU) o = o > 0.f ? o : 0.f;
                if (OUT_BF16) ((__bf16*)Outptr)[(size_t)row * N + col] = to_bf16(o);
                else          ((float*)Outptr)[(size_t)row * N + col]  = o;
            }
        }
    }
}

// ---- weight transpose + bf16 convert: in fp32 [K,N] -> out bf16 [N,K]
__global__ void k_prep_wt(const float* __restrict__ Win, __bf16* __restrict__ Wout,
                          int K, int N) {
    int idx = blockIdx.x * blockDim.x + threadIdx.x;
    if (idx >= K * N) return;
    int k = idx / N, n = idx - k * N;
    Wout[(size_t)n * K + k] = to_bf16(Win[idx]);
}

// ---- lin_pre stage 1: tmp = relu(x @ W1 + b1), stored bf16 [Nn,256]
__global__ void k_pre1(const float* __restrict__ x, const float* __restrict__ W1,
                       const float* __restrict__ b1, __bf16* __restrict__ tmp, int Nn) {
    int idx = blockIdx.x * blockDim.x + threadIdx.x;
    if (idx >= Nn * 256) return;
    int g = idx >> 8, c = idx & 255;
    float v = x[g * 3 + 0] * W1[c] + x[g * 3 + 1] * W1[256 + c]
            + x[g * 3 + 2] * W1[512 + c] + b1[c];
    tmp[idx] = to_bf16(v > 0.f ? v : 0.f);
}

__global__ void k_node_init(float* __restrict__ deg, unsigned char* __restrict__ touched, int Nn) {
    int i = blockIdx.x * blockDim.x + threadIdx.x;
    if (i < Nn) { deg[i] = 1.f; touched[i] = 0; }     // self-loop contributes 1
}

__global__ void k_deg_edges(const long long* __restrict__ ei, float* __restrict__ deg, int E) {
    int e = blockIdx.x * blockDim.x + threadIdx.x;
    if (e < E) atomicAdd(&deg[(int)ei[(size_t)E + e]], 1.f);
}

__global__ void k_dinv(const float* __restrict__ deg, float* __restrict__ dinv, int Nn) {
    int i = blockIdx.x * blockDim.x + threadIdx.x;
    if (i < Nn) dinv[i] = rsqrtf(deg[i]);
}

__global__ void k_touch(const long long* __restrict__ ei, unsigned char* __restrict__ touched, int E) {
    int e = blockIdx.x * blockDim.x + threadIdx.x;
    if (e < E) { touched[(int)ei[e]] = 1; touched[(int)ei[(size_t)E + e]] = 1; }
}

// ---- agg init: bias + self-loop term
__global__ void k_agg_init(const float* __restrict__ msg, const float* __restrict__ dinv,
                           const float* __restrict__ bvec, float* __restrict__ agg, int Nn) {
    int idx = blockIdx.x * blockDim.x + threadIdx.x;
    if (idx >= Nn * 128) return;
    int g = idx >> 7, c = idx & 127;
    float d = dinv[g];
    agg[idx] = bvec[c] + msg[idx] * d * d;
}

// ---- edge scatter: one wave per edge, 4 channels/lane (float4), fp32 atomics
__global__ __launch_bounds__(256) void k_agg_edges(const long long* __restrict__ ei,
                                                   const float* __restrict__ msg,
                                                   const float* __restrict__ dinv,
                                                   float* __restrict__ agg, int E) {
    int e = blockIdx.x * 8 + (threadIdx.x >> 5);
    if (e >= E) return;
    int lane = threadIdx.x & 31;
    int s = (int)ei[e], d = (int)ei[(size_t)E + e];
    float w = dinv[s] * dinv[d];
    float4 m = ((const float4*)(msg + (size_t)s * 128))[lane];
    float* ap = agg + (size_t)d * 128 + lane * 4;
    atomicAdd(ap + 0, m.x * w);
    atomicAdd(ap + 1, m.y * w);
    atomicAdd(ap + 2, m.z * w);
    atomicAdd(ap + 3, m.w * w);
}

__global__ void k_residual(float* __restrict__ agg, const float* __restrict__ prex, int n) {
    int i = blockIdx.x * blockDim.x + threadIdx.x;
    if (i < n) { float v = agg[i] + prex[i]; agg[i] = v > 0.f ? v : 0.f; }
}

// ---- logits + mask: one wave per (b, n); 128-dim dot, wave reduce, fused mask
__global__ __launch_bounds__(256) void k_logits(const float* __restrict__ h,
                                                const float* __restrict__ x,
                                                const float* __restrict__ cap,
                                                const long long* __restrict__ prev,
                                                const unsigned char* __restrict__ touched,
                                                float* __restrict__ out, int B) {
    int item = blockIdx.x * 8 + (threadIdx.x >> 5);
    if (item >= B * NNODES) return;
    int b = item / NNODES, n = item - b * NNODES;
    int pg = b * NNODES + (int)prev[b];
    int lane = threadIdx.x & 31;
    float4 hv = ((const float4*)(h + (size_t)item * 128))[lane];
    float4 ev = ((const float4*)(h + (size_t)pg * 128))[lane];
    float p = hv.x * ev.x + hv.y * ev.y + hv.z * ev.z + hv.w * ev.w;
    #pragma unroll
    for (int off = 16; off > 0; off >>= 1) p += __shfl_down(p, off, 32);
    if (lane == 0) {
        bool m;
        if (n == 0) m = (prev[b] == 0);
        else        m = (touched[item] != 0) || (x[(size_t)item * 3 + 2] > cap[b]);
        out[item] = m ? NEGV : p;
    }
}

extern "C" void kernel_launch(void* const* d_in, const int* in_sizes, int n_in,
                              void* d_out, int out_size, void* d_ws, size_t ws_size,
                              hipStream_t stream) {
    const float*     x       = (const float*)d_in[0];
    const float*     cap     = (const float*)d_in[1];
    const long long* ei      = (const long long*)d_in[2];
    const long long* prev    = (const long long*)d_in[3];
    const float*     pre_W1  = (const float*)d_in[4];
    const float*     pre_b1  = (const float*)d_in[5];
    const float*     pre_W2  = (const float*)d_in[6];
    const float*     pre_b2  = (const float*)d_in[7];
    const float*     conv_W  = (const float*)d_in[8];
    const float*     conv_b  = (const float*)d_in[9];
    const float*     bn_g    = (const float*)d_in[10];
    const float*     bn_b    = (const float*)d_in[11];
    const float*     bn_m    = (const float*)d_in[12];
    const float*     bn_v    = (const float*)d_in[13];
    const float*     post_W1 = (const float*)d_in[14];
    const float*     post_b1 = (const float*)d_in[15];
    const float*     post_W2 = (const float*)d_in[16];
    const float*     post_b2 = (const float*)d_in[17];

    const int Nn = in_sizes[0] / 3;
    const int B  = in_sizes[1];
    const int E  = in_sizes[2] / 2;

    char* cur = (char*)d_ws;
    auto alloc = [&](size_t bytes) -> void* {
        void* p = (void*)cur;
        cur += (bytes + 255) & ~(size_t)255;
        return p;
    };
    __bf16*        tmp_bf   = (__bf16*)alloc((size_t)Nn * 256 * 2);
    float*         pre_x    = (float*)alloc((size_t)Nn * 128 * 4);
    float*         msg      = (float*)alloc((size_t)Nn * 128 * 4);
    float*         agg      = (float*)alloc((size_t)Nn * 128 * 4);
    float*         deg      = (float*)alloc((size_t)Nn * 4);
    float*         dinv     = (float*)alloc((size_t)Nn * 4);
    unsigned char* touched  = (unsigned char*)alloc((size_t)Nn);
    __bf16*        wt_pre2  = (__bf16*)alloc((size_t)128 * 256 * 2);
    __bf16*        wt_conv  = (__bf16*)alloc((size_t)4 * 128 * 128 * 2);
    __bf16*        wt_post1 = (__bf16*)alloc((size_t)256 * 128 * 2);
    __bf16*        wt_post2 = (__bf16*)alloc((size_t)128 * 256 * 2);

    const int TB = 256;
    auto cdiv = [](int a, int b) { return (a + b - 1) / b; };

    // ---- weight prep (bf16, [N][K])
    k_prep_wt<<<cdiv(256 * 128, TB), TB, 0, stream>>>(pre_W2, wt_pre2, 256, 128);
    for (int i = 0; i < 4; ++i)
        k_prep_wt<<<cdiv(128 * 128, TB), TB, 0, stream>>>(conv_W + (size_t)i * 128 * 128,
                                                          wt_conv + (size_t)i * 128 * 128, 128, 128);
    k_prep_wt<<<cdiv(128 * 256, TB), TB, 0, stream>>>(post_W1, wt_post1, 128, 256);
    k_prep_wt<<<cdiv(256 * 128, TB), TB, 0, stream>>>(post_W2, wt_post2, 256, 128);

    // ---- degree / norm / touch mask
    k_node_init<<<cdiv(Nn, TB), TB, 0, stream>>>(deg, touched, Nn);
    k_deg_edges<<<cdiv(E, TB), TB, 0, stream>>>(ei, deg, E);
    k_dinv<<<cdiv(Nn, TB), TB, 0, stream>>>(deg, dinv, Nn);
    k_touch<<<cdiv(E, TB), TB, 0, stream>>>(ei, touched, E);

    // ---- lin_pre
    k_pre1<<<cdiv(Nn * 256, TB), TB, 0, stream>>>(x, pre_W1, pre_b1, tmp_bf, Nn);
    gemm_wmma_bf16<true, false, false, false, false>
        <<<dim3(cdiv(Nn, 128), 1), TB, 0, stream>>>(
            tmp_bf, wt_pre2, pre_b2, nullptr, nullptr, nullptr, nullptr,
            pre_x, Nn, 256, 128);

    // ---- 4 GCN layers (BN+ReLU of layer i-1 fused into layer i's GEMM input)
    for (int i = 0; i < 4; ++i) {
        if (i == 0) {
            gemm_wmma_bf16<false, false, false, false, false>
                <<<dim3(cdiv(Nn, 128), 1), TB, 0, stream>>>(
                    pre_x, wt_conv, nullptr, nullptr, nullptr, nullptr, nullptr,
                    msg, Nn, 128, 128);
        } else {
            gemm_wmma_bf16<false, true, true, false, false>
                <<<dim3(cdiv(Nn, 128), 1), TB, 0, stream>>>(
                    agg, wt_conv + (size_t)i * 128 * 128, nullptr,
                    bn_g + (size_t)(i - 1) * 128, bn_b + (size_t)(i - 1) * 128,
                    bn_m + (size_t)(i - 1) * 128, bn_v + (size_t)(i - 1) * 128,
                    msg, Nn, 128, 128);
        }
        k_agg_init<<<cdiv(Nn * 128, TB), TB, 0, stream>>>(msg, dinv, conv_b + (size_t)i * 128, agg, Nn);
        k_agg_edges<<<cdiv(E, 8), TB, 0, stream>>>(ei, msg, dinv, agg, E);
    }

    // ---- residual + ReLU
    k_residual<<<cdiv(Nn * 128, TB), TB, 0, stream>>>(agg, pre_x, Nn * 128);

    // ---- lin_post (stage1 out relu'd to bf16; stage2 back to fp32 in msg)
    gemm_wmma_bf16<false, false, false, true, true>
        <<<dim3(cdiv(Nn, 128), 2), TB, 0, stream>>>(
            agg, wt_post1, post_b1, nullptr, nullptr, nullptr, nullptr,
            tmp_bf, Nn, 128, 256);
    gemm_wmma_bf16<true, false, false, false, false>
        <<<dim3(cdiv(Nn, 128), 1), TB, 0, stream>>>(
            tmp_bf, wt_post2, post_b2, nullptr, nullptr, nullptr, nullptr,
            msg, Nn, 256, 128);

    // ---- attention logits + mask
    k_logits<<<cdiv(B * NNODES, 8), TB, 0, stream>>>(msg, x, cap, prev, touched,
                                                     (float*)d_out, B);
}
